// Block_14912126452368
// MI455X (gfx1250) — matrix operand order repaired and checked by reading, not compiled, not run on hardware
//
#include <hip/hip_runtime.h>

// ---------------------------------------------------------------------------
// CDNA5 (gfx1250) transformer block: LN1 -> QKV -> MHA -> Wo+res -> LN2 ->
// GELU-MLP -> +res.  All GEMMs via v_wmma_f32_16x16x32_bf16 (wave32).
// ---------------------------------------------------------------------------

typedef __attribute__((ext_vector_type(16))) __bf16 v16bf;
typedef __attribute__((ext_vector_type(8)))  float  v8f;

#define DIM_C    1024
#define HIDDEN_C 4096
#define HEADS_C  16
#define HD_C     64
#define BATCH_C  16
#define NSEQ_C   1024
#define MROWS_C  (BATCH_C * NSEQ_C)   // 16384

union FragBF { v16bf v; unsigned int u[8]; __bf16 e[16]; };

// ---------------------------------------------------------------------------
// fp32 [R,C] -> bf16 transposed [C,R]   (weights, done once per launch)
// ---------------------------------------------------------------------------
__global__ void k_transpose_bf16(const float* __restrict__ in,
                                 __bf16* __restrict__ out, int R, int C) {
  int idx = blockIdx.x * blockDim.x + threadIdx.x;
  int r = idx / C, c = idx % C;
  out[(size_t)c * R + r] = (__bf16)in[(size_t)r * C + c];
}

// ---------------------------------------------------------------------------
// LayerNorm (fp32 in) -> bf16 out.  One row (DIM=1024) per 256-thread block.
// ---------------------------------------------------------------------------
__global__ void k_layernorm_bf16(const float* __restrict__ x,
                                 const float* __restrict__ g,
                                 const float* __restrict__ b,
                                 __bf16* __restrict__ out) {
  const int row = blockIdx.x;
  const float* xr = x + (size_t)row * DIM_C;
  float vals[4], lsum = 0.f, lsq = 0.f;
#pragma unroll
  for (int i = 0; i < 4; ++i) {
    float v = xr[threadIdx.x + i * 256];
    vals[i] = v; lsum += v; lsq += v * v;
  }
#pragma unroll
  for (int o = 16; o > 0; o >>= 1) {
    lsum += __shfl_xor(lsum, o, 32);
    lsq  += __shfl_xor(lsq,  o, 32);
  }
  __shared__ float ssum[8], ssq[8];
  if ((threadIdx.x & 31) == 0) { ssum[threadIdx.x >> 5] = lsum; ssq[threadIdx.x >> 5] = lsq; }
  __syncthreads();
  float ts = 0.f, tq = 0.f;
#pragma unroll
  for (int i = 0; i < 8; ++i) { ts += ssum[i]; tq += ssq[i]; }
  const float mean = ts * (1.f / DIM_C);
  const float var  = tq * (1.f / DIM_C) - mean * mean;
  const float rstd = rsqrtf(var + 1e-5f);
#pragma unroll
  for (int i = 0; i < 4; ++i) {
    int c = threadIdx.x + i * 256;
    out[(size_t)row * DIM_C + c] = (__bf16)((vals[i] - mean) * rstd * g[c] + b[c]);
  }
}

// ---------------------------------------------------------------------------
// Generic WMMA GEMM:  out[M,Nc] = act(A[M,K]_bf16 @ Wt[Nc,K]^T + bias (+res))
// Block: 256 threads = 8 waves (2 row-waves x 4 col-waves) -> 64 x 256 tile.
// Each wave: 32x64 tile = 2 A-frags x 4 B-frags = 8 WMMA per K-step
// (12 b128 loads per 8 WMMA).  act: 0 = none, 1 = exact GELU.
// Exactly one of outF / outB is non-null.
// ---------------------------------------------------------------------------
__global__ void k_gemm_wmma(const __bf16* __restrict__ A,
                            const __bf16* __restrict__ Wt,
                            const float* __restrict__ bias,
                            const float* __restrict__ residual,
                            float* __restrict__ outF,
                            __bf16* __restrict__ outB,
                            int M, int K, int Nc, int act) {
  const int lane = threadIdx.x & 31;
  const int wid  = threadIdx.x >> 5;
  const int wrow = wid >> 2;          // 0..1
  const int wcol = wid & 3;           // 0..3
  const int half = lane >> 4;
  const int ln   = lane & 15;
  const int rowBase = blockIdx.y * 64 + wrow * 32;
  const int colBase = blockIdx.x * 256 + wcol * 64;

  v8f acc[2][4];
#pragma unroll
  for (int rt = 0; rt < 2; ++rt)
#pragma unroll
    for (int t = 0; t < 4; ++t) acc[rt][t] = (v8f)0.f;

  const __bf16* Arow0 = A + (size_t)(rowBase + ln) * K;
  const __bf16* Arow1 = A + (size_t)(rowBase + 16 + ln) * K;
  const __bf16* Wrow[4];
#pragma unroll
  for (int t = 0; t < 4; ++t) Wrow[t] = Wt + (size_t)(colBase + t * 16 + ln) * K;

  for (int k0 = 0; k0 < K; k0 += 32) {
    // speculative near-cache prefetch of next K-step (safe past end)
    __builtin_prefetch(Arow0 + k0 + 32, 0, 3);    // global_prefetch_b8, WGP scope
    __builtin_prefetch(Wrow[0] + k0 + 32, 0, 3);

    FragBF a0, a1;
#pragma unroll
    for (int p = 0; p < 8; ++p) {
      int k = k0 + 2 * (p & 3) + 16 * (p >> 2) + 8 * half;   // 16-bit A layout
      a0.u[p] = *(const unsigned int*)(Arow0 + k);
      a1.u[p] = *(const unsigned int*)(Arow1 + k);
    }
#pragma unroll
    for (int t = 0; t < 4; ++t) {
      FragBF bfr;
#pragma unroll
      for (int p = 0; p < 8; ++p) {
        int k = k0 + 2 * (p & 3) + 16 * (p >> 2) + 8 * half;  // 16-bit B layout
        bfr.u[p] = *(const unsigned int*)(Wrow[t] + k);
      }
      acc[0][t] = __builtin_amdgcn_wmma_f32_16x16x32_bf16(
          false, a0.v, false, bfr.v, (short)0, acc[0][t], false, false);
      acc[1][t] = __builtin_amdgcn_wmma_f32_16x16x32_bf16(
          false, a1.v, false, bfr.v, (short)0, acc[1][t], false, false);
    }
  }

#pragma unroll
  for (int rt = 0; rt < 2; ++rt)
#pragma unroll
    for (int t = 0; t < 4; ++t) {
      const int n = colBase + t * 16 + ln;
      const float bv = bias ? bias[n] : 0.f;
#pragma unroll
      for (int r = 0; r < 8; ++r) {
        const int m = rowBase + rt * 16 + r + 8 * half;       // C/D layout
        float v = acc[rt][t][r] + bv;
        if (residual) v += residual[(size_t)m * Nc + n];
        if (act == 1) v = 0.5f * v * (1.0f + erff(v * 0.70710678118654752f));
        if (outF) outF[(size_t)m * Nc + n] = v;
        else      outB[(size_t)m * Nc + n] = (__bf16)v;
      }
    }
}

// ---------------------------------------------------------------------------
// Flash attention.  grid = (N/64, B*HEADS), block = 128 (4 waves).
// Each wave owns a 16-query tile.  Per 32-key chunk:
//   S^T tiles  T0,T1 = K . Q^T  (WMMA, C-layout column = query lane)
//   online softmax per column, then P^T lands in B-layout (one half-swap)
//   O^T += V^T . P^T  (WMMA), V chunk staged in LDS once for all 4 waves.
// ---------------------------------------------------------------------------
__global__ void k_flash_attn(const __bf16* __restrict__ Q,
                             const __bf16* __restrict__ K,
                             const __bf16* __restrict__ V,
                             __bf16* __restrict__ O) {
  const int bh   = blockIdx.y;
  const int b    = bh / HEADS_C;
  const int h    = bh % HEADS_C;
  const int wid  = threadIdx.x >> 5;
  const int lane = threadIdx.x & 31;
  const int half = lane >> 4;
  const int ln   = lane & 15;
  const int qbase = blockIdx.x * 64 + wid * 16;
  const size_t bN  = (size_t)b * NSEQ_C;
  const size_t h64 = (size_t)h * HD_C;
  const float scale = 0.125f;  // HEAD_DIM^-0.5

  __shared__ __align__(16) __bf16 Vs[32 * HD_C];   // 32 keys x 64 dims = 4KB

  // Q^T B-fragments (d segments 0..31, 32..63) -- loaded once
  FragBF qf[2];
#pragma unroll
  for (int s = 0; s < 2; ++s)
#pragma unroll
    for (int p = 0; p < 8; ++p) {
      int d = s * 32 + 2 * (p & 3) + 16 * (p >> 2) + 8 * half;
      qf[s].u[p] = *(const unsigned int*)(Q + (bN + qbase + ln) * DIM_C + h64 + d);
    }

  v8f oacc[4];
#pragma unroll
  for (int t = 0; t < 4; ++t) oacc[t] = (v8f)0.f;
  float mstat = -__builtin_inff();
  float lstat = 0.f;

  for (int j = 0; j < NSEQ_C / 32; ++j) {
    const int keybase = j * 32;

    // ---- cooperative V stage: 32 rows x 128B, uint4 loads ----
    __syncthreads();
#pragma unroll
    for (int it = 0; it < 2; ++it) {
      int e   = threadIdx.x + it * 128;   // 0..255
      int row = e >> 3, seg = e & 7;
      ((uint4*)Vs)[e] =
          ((const uint4*)(V + (bN + keybase + row) * DIM_C + h64))[seg];
    }
    __syncthreads();

    // ---- S^T tiles: T = K_chunk . Q^T ----
    v8f s0 = (v8f)0.f, s1 = (v8f)0.f;
#pragma unroll
    for (int s = 0; s < 2; ++s) {
      FragBF ka0, ka1;
#pragma unroll
      for (int p = 0; p < 8; ++p) {
        int d = s * 32 + 2 * (p & 3) + 16 * (p >> 2) + 8 * half;
        ka0.u[p] = *(const unsigned int*)(K + (bN + keybase + ln)      * DIM_C + h64 + d);
        ka1.u[p] = *(const unsigned int*)(K + (bN + keybase + 16 + ln) * DIM_C + h64 + d);
      }
      s0 = __builtin_amdgcn_wmma_f32_16x16x32_bf16(false, ka0.v, false, qf[s].v,
                                                   (short)0, s0, false, false);
      s1 = __builtin_amdgcn_wmma_f32_16x16x32_bf16(false, ka1.v, false, qf[s].v,
                                                   (short)0, s1, false, false);
    }

    // ---- online softmax over this chunk (per query column = per lane) ----
    float p0[8], p1[8];
    float cmax = -__builtin_inff();
#pragma unroll
    for (int r = 0; r < 8; ++r) {
      p0[r] = s0[r] * scale;  p1[r] = s1[r] * scale;
      cmax = fmaxf(cmax, fmaxf(p0[r], p1[r]));
    }
    cmax = fmaxf(cmax, __shfl_xor(cmax, 16, 32));
    const float newm = fmaxf(mstat, cmax);
    const float corr = __expf(mstat - newm);
    mstat = newm;
    float psum = 0.f;
#pragma unroll
    for (int r = 0; r < 8; ++r) {
      p0[r] = __expf(p0[r] - newm);
      p1[r] = __expf(p1[r] - newm);
      psum += p0[r] + p1[r];
    }
    psum += __shfl_xor(psum, 16, 32);
    lstat = lstat * corr + psum;
#pragma unroll
    for (int t = 0; t < 4; ++t)
#pragma unroll
      for (int r = 0; r < 8; ++r) oacc[t][r] *= corr;

    // ---- build P^T B-fragment (half-swap via shfl_xor 16) ----
    float sh0[8], sh1[8];
#pragma unroll
    for (int r = 0; r < 8; ++r) {
      sh0[r] = __shfl_xor(p0[r], 16, 32);
      sh1[r] = __shfl_xor(p1[r], 16, 32);
    }
    FragBF pf;
#pragma unroll
    for (int i = 0; i < 8; ++i) {
      pf.e[i]     = (__bf16)(half ? sh1[i] : p0[i]);   // kk = i      (+16*half)
      pf.e[i + 8] = (__bf16)(half ? p1[i]  : sh0[i]);  // kk = i + 8  (+16*half)
    }

    // ---- O^T += V^T . P^T ----
#pragma unroll
    for (int dt = 0; dt < 4; ++dt) {
      FragBF va;
#pragma unroll
      for (int p = 0; p < 8; ++p) {
        int kk = 2 * p + 16 * half;
        va.e[2 * p]     = Vs[kk * HD_C + dt * 16 + ln];
        va.e[2 * p + 1] = Vs[(kk + 1) * HD_C + dt * 16 + ln];
      }
      oacc[dt] = __builtin_amdgcn_wmma_f32_16x16x32_bf16(
          false, va.v, false, pf.v, (short)0, oacc[dt], false, false);
    }
  }

  // ---- normalize and store O (transposed accumulators) ----
  const float inv = 1.0f / lstat;
#pragma unroll
  for (int dt = 0; dt < 4; ++dt)
#pragma unroll
    for (int r = 0; r < 8; ++r) {
      int drow = dt * 16 + r + 8 * half;
      O[(bN + qbase + ln) * DIM_C + h64 + drow] = (__bf16)(oacc[dt][r] * inv);
    }
}

// ---------------------------------------------------------------------------
// Host orchestration.  Workspace layout (bytes, 248 MB total):
//   [ 0,  24M) bf16 transposed weights  wq,wk,wv,wo (2M ea) | w1 (8M) | w2 (8M)
//   [24M, 56M) h1 / h2 bf16 activations (32M)
//   [56M,184M) q(32M) k(32M) v(32M) attn(32M)  -- reused as MLP hidden (128M)
//   [184M,248M) x2 fp32 (64M)
// ---------------------------------------------------------------------------
extern "C" void kernel_launch(void* const* d_in, const int* in_sizes, int n_in,
                              void* d_out, int out_size, void* d_ws, size_t ws_size,
                              hipStream_t stream) {
  (void)in_sizes; (void)n_in; (void)out_size; (void)ws_size;
  const float* x     = (const float*)d_in[0];
  const float* ln1_g = (const float*)d_in[1];
  const float* ln1_b = (const float*)d_in[2];
  const float* wq    = (const float*)d_in[3];
  const float* wk    = (const float*)d_in[4];
  const float* wv    = (const float*)d_in[5];
  const float* wo    = (const float*)d_in[6];
  const float* bo    = (const float*)d_in[7];
  const float* ln2_g = (const float*)d_in[8];
  const float* ln2_b = (const float*)d_in[9];
  const float* w1    = (const float*)d_in[10];
  const float* b1    = (const float*)d_in[11];
  const float* w2    = (const float*)d_in[12];
  const float* b2    = (const float*)d_in[13];

  char* ws = (char*)d_ws;
  const size_t MB = 1u << 20;
  __bf16* wq_t = (__bf16*)(ws + 0 * MB);
  __bf16* wk_t = (__bf16*)(ws + 2 * MB);
  __bf16* wv_t = (__bf16*)(ws + 4 * MB);
  __bf16* wo_t = (__bf16*)(ws + 6 * MB);
  __bf16* w1_t = (__bf16*)(ws + 8 * MB);
  __bf16* w2_t = (__bf16*)(ws + 16 * MB);
  __bf16* h1   = (__bf16*)(ws + 24 * MB);   // also h2
  __bf16* qb   = (__bf16*)(ws + 56 * MB);
  __bf16* kb   = (__bf16*)(ws + 88 * MB);
  __bf16* vb   = (__bf16*)(ws + 120 * MB);
  __bf16* attn = (__bf16*)(ws + 152 * MB);
  __bf16* hid  = (__bf16*)(ws + 56 * MB);   // 128 MB, reuses q/k/v/attn
  float*  x2   = (float*)(ws + 184 * MB);

  const int M = MROWS_C;

  // 1. weight convert + transpose to bf16 [Nc, K]
  k_transpose_bf16<<<(DIM_C * DIM_C) / 256, 256, 0, stream>>>(wq, wq_t, DIM_C, DIM_C);
  k_transpose_bf16<<<(DIM_C * DIM_C) / 256, 256, 0, stream>>>(wk, wk_t, DIM_C, DIM_C);
  k_transpose_bf16<<<(DIM_C * DIM_C) / 256, 256, 0, stream>>>(wv, wv_t, DIM_C, DIM_C);
  k_transpose_bf16<<<(DIM_C * DIM_C) / 256, 256, 0, stream>>>(wo, wo_t, DIM_C, DIM_C);
  k_transpose_bf16<<<(DIM_C * HIDDEN_C) / 256, 256, 0, stream>>>(w1, w1_t, DIM_C, HIDDEN_C);
  k_transpose_bf16<<<(HIDDEN_C * DIM_C) / 256, 256, 0, stream>>>(w2, w2_t, HIDDEN_C, DIM_C);

  // 2. LN1
  k_layernorm_bf16<<<M, 256, 0, stream>>>(x, ln1_g, ln1_b, h1);

  // 3. Q, K, V projections
  dim3 gD(DIM_C / 256, M / 64);
  k_gemm_wmma<<<gD, 256, 0, stream>>>(h1, wq_t, nullptr, nullptr, nullptr, qb, M, DIM_C, DIM_C, 0);
  k_gemm_wmma<<<gD, 256, 0, stream>>>(h1, wk_t, nullptr, nullptr, nullptr, kb, M, DIM_C, DIM_C, 0);
  k_gemm_wmma<<<gD, 256, 0, stream>>>(h1, wv_t, nullptr, nullptr, nullptr, vb, M, DIM_C, DIM_C, 0);

  // 4. multi-head flash attention
  k_flash_attn<<<dim3(NSEQ_C / 64, BATCH_C * HEADS_C), 128, 0, stream>>>(qb, kb, vb, attn);

  // 5. x2 = attn @ wo + bo + x
  k_gemm_wmma<<<gD, 256, 0, stream>>>(attn, wo_t, bo, x, x2, nullptr, M, DIM_C, DIM_C, 0);

  // 6. LN2
  k_layernorm_bf16<<<M, 256, 0, stream>>>(x2, ln2_g, ln2_b, h1);

  // 7. hidden = gelu(h2 @ w1 + b1)
  dim3 gH(HIDDEN_C / 256, M / 64);
  k_gemm_wmma<<<gH, 256, 0, stream>>>(h1, w1_t, b1, nullptr, nullptr, hid, M, DIM_C, HIDDEN_C, 1);

  // 8. out = hidden @ w2 + b2 + x2
  k_gemm_wmma<<<gD, 256, 0, stream>>>(hid, w2_t, b2, x2, (float*)d_out, nullptr, M, HIDDEN_C, DIM_C, 0);
}
